// SelfMemoryEncode_40630390620997
// MI455X (gfx1250) — matrix operand order repaired and checked by reading, not compiled, not run on hardware
//
#include <hip/hip_runtime.h>
#include <hip/hip_bf16.h>

typedef __bf16 bf16_t;
typedef __attribute__((ext_vector_type(8)))  bf16_t bf16x8;
typedef __attribute__((ext_vector_type(16))) bf16_t bf16x16;
typedef __attribute__((ext_vector_type(8)))  float  f32x8;

#define D_  1024
#define H_  16
#define M_  1024
#define HD_ 64
#define VD_ 16
#define B_  2
#define S_  1024
#define W_  1021   // (1024 - 4)/1 + 1

union Frag16 { bf16x16 v; bf16x8 h[2]; };

// Single-tile K-loop (used by the attention kernel's short GEMMs).
__device__ __forceinline__ f32x8 tile_mma(const bf16_t* __restrict__ Arow,
                                          const bf16_t* __restrict__ Brow,
                                          int K, int hf, f32x8 acc) {
  for (int k0 = 0; k0 < K; k0 += 32) {
    Frag16 a, b;
    const bf16_t* ap = Arow + k0 + hf * 8;
    a.h[0] = *(const bf16x8*)(ap);
    a.h[1] = *(const bf16x8*)(ap + 16);
    const bf16_t* bp = Brow + k0 + hf * 8;
    b.h[0] = *(const bf16x8*)(bp);
    b.h[1] = *(const bf16x8*)(bp + 16);
    acc = __builtin_amdgcn_wmma_f32_16x16x32_bf16(false, a.v, false, b.v,
                                                  (short)0, acc, false, false);
  }
  return acc;
}

__global__ void sme_cvt_kernel(const float* __restrict__ s,
                               bf16_t* __restrict__ d, int n) {
  int i = blockIdx.x * blockDim.x + threadIdx.x;
  int stride = gridDim.x * blockDim.x;
  for (; i < n; i += stride) d[i] = (bf16_t)s[i];
}

// ---------------------------------------------------------------------------
// 2x4 register-blocked GEMM body: 32(M) x 64(N) per wave, K in steps of 32.
// 12 b128 loads feed 8 v_wmma per K-step (24B/lane per WMMA vs 64 for 1x1).
// ---------------------------------------------------------------------------
__device__ __forceinline__ void mma_2x4(const bf16_t* __restrict__ Ar0,
                                        const bf16_t* __restrict__ Ar1,
                                        const bf16_t* __restrict__ Br0,
                                        const bf16_t* __restrict__ Br1,
                                        const bf16_t* __restrict__ Br2,
                                        const bf16_t* __restrict__ Br3,
                                        int K, int hf, f32x8 acc[2][4]) {
  const bf16_t* Ar[2] = {Ar0, Ar1};
  const bf16_t* Br[4] = {Br0, Br1, Br2, Br3};
  for (int k0 = 0; k0 < K; k0 += 32) {
    Frag16 a[2], b[4];
#pragma unroll
    for (int i = 0; i < 2; ++i) {
      const bf16_t* ap = Ar[i] + k0 + hf * 8;
      a[i].h[0] = *(const bf16x8*)(ap);
      a[i].h[1] = *(const bf16x8*)(ap + 16);
    }
#pragma unroll
    for (int j = 0; j < 4; ++j) {
      const bf16_t* bp = Br[j] + k0 + hf * 8;
      b[j].h[0] = *(const bf16x8*)(bp);
      b[j].h[1] = *(const bf16x8*)(bp + 16);
      if (k0 + 32 < K) __builtin_prefetch(Br[j] + k0 + 64, 0, 0);
    }
#pragma unroll
    for (int i = 0; i < 2; ++i)
#pragma unroll
      for (int j = 0; j < 4; ++j)
        acc[i][j] = __builtin_amdgcn_wmma_f32_16x16x32_bf16(
            false, a[i].v, false, b[j].v, (short)0, acc[i][j], false, false);
  }
}

// q = x @ Wq^T + bq  (2048 x 1024 x 1024), output bf16
__global__ __launch_bounds__(256)
void sme_qgemm_kernel(const bf16_t* __restrict__ xb,
                      const bf16_t* __restrict__ Wqb,
                      const float*  __restrict__ bq,
                      bf16_t* __restrict__ qb) {
  const int tid  = threadIdx.x;
  const int lane = tid & 31;
  const int cn16 = lane & 15;
  const int hf   = lane >> 4;
  const int wid  = blockIdx.x * 8 + (tid >> 5);   // 1024 waves
  const int mt   = wid >> 4;                      // 0..63  (rows of 32)
  const int nt   = wid & 15;                      // 0..15  (cols of 64)
  const int m0   = mt * 32, n0 = nt * 64;

  f32x8 acc[2][4] = {};
  mma_2x4(xb + (long)(m0 + cn16) * D_,
          xb + (long)(m0 + 16 + cn16) * D_,
          Wqb + (long)(n0 + cn16) * D_,
          Wqb + (long)(n0 + 16 + cn16) * D_,
          Wqb + (long)(n0 + 32 + cn16) * D_,
          Wqb + (long)(n0 + 48 + cn16) * D_,
          D_, hf, acc);

#pragma unroll
  for (int i = 0; i < 2; ++i)
#pragma unroll
    for (int j = 0; j < 4; ++j) {
      const int cn = n0 + j * 16 + cn16;
      const float bias = bq[cn];
#pragma unroll
      for (int r = 0; r < 8; ++r)
        qb[(long)(m0 + i * 16 + hf * 8 + r) * D_ + cn] =
            (bf16_t)(acc[i][j][r] + bias);
    }
}

// Per (b,t): logits = q_heads(16x64) @ Wk^T -> softmax -> V = att @ Wv^T (16x16)
__global__ __launch_bounds__(128)
void sme_attn_kernel(const bf16_t* __restrict__ qb,
                     const bf16_t* __restrict__ Wkb,
                     const float*  __restrict__ bk,
                     const bf16_t* __restrict__ Wvb,
                     const float*  __restrict__ bv,
                     bf16_t* __restrict__ Vbf) {
  __shared__ float lg[H_][M_];                      // 64 KB logits / exp scratch
  __shared__ __align__(16) bf16_t attb[H_][M_];     // 32 KB bf16 attention
  __shared__ float red[H_][8];
  __shared__ float stat[H_];
  __shared__ float pc[4][8][32];                    // split-K partial C

  const int tid  = threadIdx.x;
  const int wave = tid >> 5;
  const int lane = tid & 31;
  const int cn16 = lane & 15;
  const int hf   = lane >> 4;
  const int bt   = blockIdx.x;                      // 0..2047

  // phase A: logits (each wave owns 256 columns = 16 tiles, K = HD = 64)
  const bf16_t* Arow = qb + (long)bt * D_ + cn16 * HD_;
  for (int j = 0; j < 16; ++j) {
    const int n0 = wave * 256 + j * 16;
    const bf16_t* Brow = Wkb + (long)(n0 + cn16) * HD_;
    f32x8 acc = {};
    acc = tile_mma(Arow, Brow, HD_, hf, acc);
    const int cn = n0 + cn16;
    const float bias = bk[cn];
#pragma unroll
    for (int r = 0; r < 8; ++r)
      lg[hf * 8 + r][cn] = (acc[r] + bias) * 0.125f;  // 1/sqrt(64)
  }
  __syncthreads();

  // phase B: row softmax over M=1024 (8 threads per row)
  const int row = tid >> 3;
  const int sub = tid & 7;
  float mx = -3.4e38f;
  for (int c = sub; c < M_; c += 8) mx = fmaxf(mx, lg[row][c]);
  red[row][sub] = mx;
  __syncthreads();
  if (sub == 0) {
    float m2 = red[row][0];
    for (int i = 1; i < 8; ++i) m2 = fmaxf(m2, red[row][i]);
    stat[row] = m2;
  }
  __syncthreads();
  const float rmax = stat[row];
  float sum = 0.f;
  for (int c = sub; c < M_; c += 8) {
    float e = __expf(lg[row][c] - rmax);
    lg[row][c] = e;
    sum += e;
  }
  red[row][sub] = sum;
  __syncthreads();
  if (sub == 0) {
    float s2 = 0.f;
    for (int i = 0; i < 8; ++i) s2 += red[row][i];
    stat[row] = 1.0f / s2;
  }
  __syncthreads();
  const float inv = stat[row];
  for (int c = sub; c < M_; c += 8) attb[row][c] = (bf16_t)(lg[row][c] * inv);
  __syncthreads();

  // phase C: V(16x16) = att(16x1024) @ Wv^T, K split 4 ways across waves
  {
    const int kbase = wave * 256;
    const bf16_t* Ar = &attb[cn16][kbase];
    const bf16_t* Br = Wvb + (long)cn16 * M_ + kbase;
    f32x8 acc = {};
    acc = tile_mma(Ar, Br, 256, hf, acc);
#pragma unroll
    for (int r = 0; r < 8; ++r) pc[wave][r][lane] = acc[r];
  }
  __syncthreads();
  {
    const int r0 = tid >> 5;                        // 0..3
    const int ln = tid & 31;
    const int c2 = ln & 15, h2 = ln >> 4;
    for (int rr = r0; rr < 8; rr += 4) {
      float v = pc[0][rr][ln] + pc[1][rr][ln] + pc[2][rr][ln] + pc[3][rr][ln];
      const int m = h2 * 8 + rr;                    // head index
      Vbf[(long)bt * (H_ * VD_) + m * VD_ + c2] = (bf16_t)(v + bv[c2]);
    }
  }
}

// out[b,w,:] = Vflat[b, 256w : 256w+1024] @ Wh^T + bh  (sliding-window GEMM)
__global__ __launch_bounds__(256)
void sme_outgemm_kernel(const bf16_t* __restrict__ Vbf,
                        const bf16_t* __restrict__ Whb,
                        const float*  __restrict__ bh,
                        float* __restrict__ out) {
  const int tid  = threadIdx.x;
  const int lane = tid & 31;
  const int cn16 = lane & 15;
  const int hf   = lane >> 4;
  const int wid  = blockIdx.x * 8 + (tid >> 5);    // 1024 waves
  const int b    = wid >> 9;                       // 0..1
  const int mt   = (wid >> 4) & 31;                // 0..31 (rows of 32)
  const int nt   = wid & 15;                       // 0..15 (cols of 64)
  const int w0   = mt * 32, n0 = nt * 64;

  int ar0 = w0 + cn16;      if (ar0 > W_ - 1) ar0 = W_ - 1;  // in-bounds windows
  int ar1 = w0 + 16 + cn16; if (ar1 > W_ - 1) ar1 = W_ - 1;
  const bf16_t* Abase = Vbf + (long)b * (S_ * 256);          // lda = 256 (stride-1 windows)

  f32x8 acc[2][4] = {};
  mma_2x4(Abase + (long)ar0 * 256,
          Abase + (long)ar1 * 256,
          Whb + (long)(n0 + cn16) * D_,
          Whb + (long)(n0 + 16 + cn16) * D_,
          Whb + (long)(n0 + 32 + cn16) * D_,
          Whb + (long)(n0 + 48 + cn16) * D_,
          D_, hf, acc);

#pragma unroll
  for (int i = 0; i < 2; ++i)
#pragma unroll
    for (int j = 0; j < 4; ++j) {
      const int cn = n0 + j * 16 + cn16;
      const float bias = bh[cn];
#pragma unroll
      for (int r = 0; r < 8; ++r) {
        const int wr = w0 + i * 16 + hf * 8 + r;
        if (wr < W_) out[((long)b * W_ + wr) * D_ + cn] = acc[i][j][r] + bias;
      }
    }
}

extern "C" void kernel_launch(void* const* d_in, const int* in_sizes, int n_in,
                              void* d_out, int out_size, void* d_ws, size_t ws_size,
                              hipStream_t stream) {
  const float* x  = (const float*)d_in[0];
  const float* Wq = (const float*)d_in[1];
  const float* bq = (const float*)d_in[2];
  const float* Wk = (const float*)d_in[3];
  const float* bk = (const float*)d_in[4];
  const float* Wv = (const float*)d_in[5];
  const float* bv = (const float*)d_in[6];
  const float* Wh = (const float*)d_in[7];
  const float* bh = (const float*)d_in[8];
  float* out = (float*)d_out;

  char* ws = (char*)d_ws;
  bf16_t* xb  = (bf16_t*)(ws);                                  // 4 MiB
  bf16_t* Wqb = (bf16_t*)(ws + (4u  << 20));                    // 2 MiB
  bf16_t* Whb = (bf16_t*)(ws + (6u  << 20));                    // 2 MiB
  bf16_t* Wkb = (bf16_t*)(ws + (8u  << 20));                    // 128 KiB
  bf16_t* Wvb = (bf16_t*)(ws + (8u  << 20) + (128u << 10));     // 32 KiB
  bf16_t* qb  = (bf16_t*)(ws + (8u  << 20) + (160u << 10));     // 4 MiB
  bf16_t* Vbf = (bf16_t*)(ws + (12u << 20) + (160u << 10));     // 1 MiB

  sme_cvt_kernel<<<1024, 256, 0, stream>>>(x,  xb,  B_ * S_ * D_);
  sme_cvt_kernel<<<1024, 256, 0, stream>>>(Wq, Wqb, D_ * D_);
  sme_cvt_kernel<<<64,   256, 0, stream>>>(Wk, Wkb, M_ * HD_);
  sme_cvt_kernel<<<16,   256, 0, stream>>>(Wv, Wvb, VD_ * M_);
  sme_cvt_kernel<<<1024, 256, 0, stream>>>(Wh, Whb, D_ * D_);

  sme_qgemm_kernel  <<<128,  256, 0, stream>>>(xb, Wqb, bq, qb);
  sme_attn_kernel   <<<2048, 128, 0, stream>>>(qb, Wkb, bk, Wvb, bv, Vbf);
  sme_outgemm_kernel<<<128,  256, 0, stream>>>(Vbf, Whb, bh, out);
}
